// RN_29686813950690
// MI455X (gfx1250) — matrix operand order repaired and checked by reading, not compiled, not run on hardware
//
#include <hip/hip_runtime.h>
#include <hip/hip_bf16.h>
#include <math.h>

// ---------------- CDNA5 WMMA types / helpers ----------------
typedef __attribute__((ext_vector_type(16))) __bf16 v16bf;
typedef __attribute__((ext_vector_type(8)))  float  v8f;

#define DEV __device__ __forceinline__

DEV float sigmf(float x) { return 1.0f / (1.0f + __expf(-x)); }

// A fragment: 16x32 (MxK) bf16, tile row-major in LDS, stride 32.
// lane<16: row=lane, elems {0..7}->K0..7, {8..15}->K16..23
// lane>=16: row=lane-16, elems {0..7}->K8..15, {8..15}->K24..31
DEV v16bf frag_a32(const __bf16* base, int lane) {
  const __bf16* p = base + (lane & 15) * 32 + ((lane >> 4) << 3);
  v16bf v;
#pragma unroll
  for (int j = 0; j < 8; ++j) { v[j] = p[j]; v[j + 8] = p[16 + j]; }
  return v;
}
// B fragment: 32x16 (KxN), stored as 16 cols x 32 K row-major (W[n][k] rows).
// lane<16: col=lane, elems {0..7}->K0..7, {8..15}->K16..23 ; lane>=16: col=lane-16, {0..7}->K8..15, {8..15}->K24..31
DEV v16bf frag_b32(const __bf16* base, int lane) {
  const __bf16* p = base + (lane & 15) * 32 + ((lane >> 4) << 3);
  v16bf v;
#pragma unroll
  for (int j = 0; j < 8; ++j) { v[j] = p[j]; v[j + 8] = p[16 + j]; }
  return v;
}
DEV v8f wmma_bf(v16bf a, v16bf b, v8f c) {
  return __builtin_amdgcn_wmma_f32_16x16x32_bf16(false, a, false, b,
                                                 (short)0, c, false, false);
}
// C layout (f32 16x16): elem e -> row = e + (lane>>4)*8, col = lane&15

// 8 fp32 -> 8 bf16 packed into uint4 (expects 16B/32B aligned src)
DEV uint4 cvt8(const float* src) {
  float4 a = *(const float4*)src;
  float4 b = *(const float4*)(src + 4);
  union { __bf16 h[8]; uint4 v; } u;
  u.h[0] = (__bf16)a.x; u.h[1] = (__bf16)a.y; u.h[2] = (__bf16)a.z; u.h[3] = (__bf16)a.w;
  u.h[4] = (__bf16)b.x; u.h[5] = (__bf16)b.y; u.h[6] = (__bf16)b.z; u.h[7] = (__bf16)b.w;
  return u.v;
}

// ---------------- problem constants ----------------
constexpr int ICNN = 2048, QH = 1024, QE = 300, QEP = 320;
constexpr int HIDN = 512, BB = 256, MAXN = 100, TT = 14;
constexpr int G4 = 4 * QH;          // 4096
constexpr int MROWS = TT * BB;      // 3584
constexpr int RTOT = BB * MAXN;     // 25600
constexpr int MLPK = ICNN + QH;     // 3072

// ------- prep: bf16 weights, padded bf16 q_feats (time-major), zero state ----
__global__ void k_prep(const float* __restrict__ qf,
                       const float* __restrict__ Wih, const float* __restrict__ Whh,
                       const float* __restrict__ Ww, const float* __restrict__ Wpw,
                       __bf16* __restrict__ qpadB,
                       __bf16* __restrict__ WihB, __bf16* __restrict__ WhhB,
                       __bf16* __restrict__ WcatB, __bf16* __restrict__ h0b,
                       float* __restrict__ cst, float* __restrict__ pre) {
  const int stride = gridDim.x * blockDim.x;
  const int t0 = blockIdx.x * blockDim.x + threadIdx.x;
  for (int i = t0; i < MROWS * QEP; i += stride) {        // q: [t*B+b][320] pad
    int r = i / QEP, k = i - r * QEP;
    int t = r >> 8, b = r & 255;
    qpadB[i] = (__bf16)((k < QE) ? qf[b * (TT * QE) + t * QE + k] : 0.0f);
  }
  for (int i = t0; i < G4 * QEP; i += stride) {           // W_ih pad 300->320
    int g = i / QEP, k = i - g * QEP;
    WihB[i] = (__bf16)((k < QE) ? Wih[g * QE + k] : 0.0f);
  }
  for (int i = t0; i < G4 * QH; i += stride) WhhB[i] = (__bf16)Whh[i];
  for (int i = t0; i < 2 * HIDN * MLPK; i += stride) {    // [W_w ; Wp_w]
    int c = i / MLPK, k = i - c * MLPK;
    WcatB[i] = (__bf16)((c < HIDN) ? Ww[c * MLPK + k] : Wpw[(c - HIDN) * MLPK + k]);
  }
  for (int i = t0; i < BB * QH; i += stride) { h0b[i] = (__bf16)0.0f; cst[i] = 0.0f; }
  for (int i = t0; i < RTOT; i += stride) pre[i] = 0.0f;
}

// ---------------- xg = q_feats @ W_ih^T + (b_ih+b_hh) ----------------
__global__ __launch_bounds__(256)
void k_xg(const __bf16* __restrict__ qpadB, const __bf16* __restrict__ WihB,
          const float* __restrict__ bih, const float* __restrict__ bhh,
          float* __restrict__ xg) {
  __shared__ __align__(16) __bf16 As[2][64][32];
  __shared__ __align__(16) __bf16 Bs[2][128][32];
  const int tid = threadIdx.x, lane = tid & 31, wv = tid >> 5;
  const int m0 = blockIdx.y * 64, n0 = blockIdx.x * 128;
  const int wm = (wv & 3) * 16, wn = (wv >> 2) * 64;
  // fixed per-thread staging chunks (16B each)
  const int rr = tid >> 2, k8 = (tid & 3) * 8;
  const __bf16* asrc = qpadB + (size_t)(m0 + rr) * QEP + k8;
  const __bf16* bsrc0 = WihB + (size_t)(n0 + rr) * QEP + k8;
  const __bf16* bsrc1 = WihB + (size_t)(n0 + rr + 64) * QEP + k8;
  uint4 ra, rb0, rb1;
  auto ld = [&](int k0) {
    ra  = *(const uint4*)(asrc + k0);
    rb0 = *(const uint4*)(bsrc0 + k0);
    rb1 = *(const uint4*)(bsrc1 + k0);
  };
  auto st = [&](int p) {
    *(uint4*)&As[p][rr][k8] = ra;
    *(uint4*)&Bs[p][rr][k8] = rb0;
    *(uint4*)&Bs[p][rr + 64][k8] = rb1;
  };
  v8f acc[4] = {};
  ld(0); st(0); __syncthreads();
  constexpr int NIT = QEP / 32;
  for (int it = 0; it < NIT; ++it) {
    const int p = it & 1;
    const bool more = (it + 1 < NIT);
    if (more) ld((it + 1) * 32);
    v16bf a = frag_a32(&As[p][wm][0], lane);
#pragma unroll
    for (int cb = 0; cb < 4; ++cb) {
      v16bf bfr = frag_b32(&Bs[p][wn + cb * 16][0], lane);
      acc[cb] = wmma_bf(a, bfr, acc[cb]);
    }
    if (more) st(1 - p);
    __syncthreads();
  }
#pragma unroll
  for (int cb = 0; cb < 4; ++cb)
#pragma unroll
    for (int e = 0; e < 8; ++e) {
      int row = m0 + wm + e + ((lane >> 4) << 3);
      int col = n0 + wn + cb * 16 + (lane & 15);
      xg[(size_t)row * G4 + col] = acc[cb][e] + bih[col] + bhh[col];
    }
}

// ------- one LSTM step: gates = xg[t] + h @ W_hh^T, fused elementwise -------
__global__ __launch_bounds__(256)
void k_lstm(const __bf16* __restrict__ hin, __bf16* __restrict__ hout,
            float* __restrict__ c, const __bf16* __restrict__ WhhB,
            const float* __restrict__ xgt) {
  __shared__ __align__(16) __bf16 As[2][64][32];
  __shared__ __align__(16) __bf16 Bs[2][4][32][32];
  const int tid = threadIdx.x, lane = tid & 31, wv = tid >> 5;
  const int b0 = blockIdx.y * 64, h0 = blockIdx.x * 32;
  const int wm = (wv & 3) * 16, wn = (wv >> 2) * 16;
  const int rr = tid >> 2, k8 = (tid & 3) * 8;
  const int g0 = tid >> 7;                 // 0..1 (chunk tid), +2 for chunk tid+256
  const int cc = (tid >> 2) & 31;
  const __bf16* asrc = hin + (size_t)(b0 + rr) * QH + k8;
  const __bf16* bsrc0 = WhhB + (size_t)(g0 * QH + h0 + cc) * QH + k8;
  const __bf16* bsrc1 = WhhB + (size_t)((g0 + 2) * QH + h0 + cc) * QH + k8;
  uint4 ra, rb0, rb1;
  auto ld = [&](int k0) {
    ra  = *(const uint4*)(asrc + k0);
    rb0 = *(const uint4*)(bsrc0 + k0);
    rb1 = *(const uint4*)(bsrc1 + k0);
  };
  auto st = [&](int p) {
    *(uint4*)&As[p][rr][k8] = ra;
    *(uint4*)&Bs[p][g0][cc][k8] = rb0;
    *(uint4*)&Bs[p][g0 + 2][cc][k8] = rb1;
  };
  v8f acc[4] = {};
  ld(0); st(0); __syncthreads();
  constexpr int NIT = QH / 32;
  for (int it = 0; it < NIT; ++it) {
    const int p = it & 1;
    const bool more = (it + 1 < NIT);
    if (more) ld((it + 1) * 32);
    v16bf a = frag_a32(&As[p][wm][0], lane);
#pragma unroll
    for (int g = 0; g < 4; ++g) {
      v16bf bfr = frag_b32(&Bs[p][g][wn][0], lane);
      acc[g] = wmma_bf(a, bfr, acc[g]);
    }
    if (more) st(1 - p);
    __syncthreads();
  }
#pragma unroll
  for (int e = 0; e < 8; ++e) {
    int b = b0 + wm + e + ((lane >> 4) << 3);
    int h = h0 + wn + (lane & 15);
    const float* xr = xgt + (size_t)b * G4;
    float gi = acc[0][e] + xr[h];
    float gf = acc[1][e] + xr[QH + h];
    float gg = acc[2][e] + xr[2 * QH + h];
    float go = acc[3][e] + xr[3 * QH + h];
    float cv = sigmf(gf) * c[b * QH + h] + sigmf(gi) * tanhf(gg);
    c[b * QH + h] = cv;
    hout[b * QH + h] = (__bf16)(sigmf(go) * tanhf(cv));
  }
}

// ---------------- fused gated MLP + f_w dot (partial) ----------------
union MlpSmem {
  struct { __bf16 A[2][64][32]; __bf16 Bm[2][256][32]; } st;  // 40KB
  float red[64][132];                                          // 33KB (padded)
};
__global__ __launch_bounds__(256)
void k_mlp(const float* __restrict__ box, const __bf16* __restrict__ hbf,
           const __bf16* __restrict__ WcatB, const float* __restrict__ Wb,
           const float* __restrict__ Wpb, const float* __restrict__ fw,
           float* __restrict__ pre) {
  __shared__ __align__(16) MlpSmem sm;
  const int tid = threadIdx.x, lane = tid & 31, wv = tid >> 5;
  const int r0 = blockIdx.y * 64, h0 = blockIdx.x * 128;
  const int wm = (wv & 3) * 16, wcg = (wv >> 2) * 64;
  // A staging: thread owns row rr, k-offset k8 (per-K-block uniform seam @2048)
  const int rr = tid >> 2, k8 = (tid & 3) * 8;
  const int r = r0 + rr;
  const float*  boxrow = box + (size_t)r * ICNN + k8;
  const __bf16* hrow   = hbf + (size_t)(r / MAXN) * QH + k8;
  // B staging: thread owns one of 256 weight rows, 64B (4x16B chunks)
  const int wrow = (tid < 128) ? (h0 + tid) : (HIDN + h0 + (tid - 128));
  const __bf16* bsrc = WcatB + (size_t)wrow * MLPK;
  uint4 ra, rb[4];
  auto ld = [&](int k0) {
    if (k0 < ICNN) ra = cvt8(boxrow + k0);
    else           ra = *(const uint4*)(hrow + (k0 - ICNN));
#pragma unroll
    for (int j = 0; j < 4; ++j) rb[j] = *(const uint4*)(bsrc + k0 + j * 8);
  };
  auto st = [&](int p) {
    *(uint4*)&sm.st.A[p][rr][k8] = ra;
#pragma unroll
    for (int j = 0; j < 4; ++j) *(uint4*)&sm.st.Bm[p][tid][j * 8] = rb[j];
  };
  v8f accY[4] = {}, accG[4] = {};
  ld(0); st(0); __syncthreads();
  constexpr int NIT = MLPK / 32;
  for (int it = 0; it < NIT; ++it) {
    const int p = it & 1;
    const bool more = (it + 1 < NIT);
    if (more) ld((it + 1) * 32);
    v16bf a = frag_a32(&sm.st.A[p][wm][0], lane);
#pragma unroll
    for (int cb = 0; cb < 4; ++cb) {
      v16bf by = frag_b32(&sm.st.Bm[p][wcg + cb * 16][0], lane);
      accY[cb] = wmma_bf(a, by, accY[cb]);
      v16bf bg = frag_b32(&sm.st.Bm[p][128 + wcg + cb * 16][0], lane);
      accG[cb] = wmma_bf(a, bg, accG[cb]);
    }
    if (more) st(1 - p);
    __syncthreads();
  }
  // si = tanh(y+Wb)*sigmoid(g+Wpb), pre-scaled by f_w; stash for row reduce
#pragma unroll
  for (int cb = 0; cb < 4; ++cb)
#pragma unroll
    for (int e = 0; e < 8; ++e) {
      int rl = wm + e + ((lane >> 4) << 3);
      int cl = wcg + cb * 16 + (lane & 15);
      int hc = h0 + cl;
      float y = tanhf(accY[cb][e] + Wb[hc]);
      float g = sigmf(accG[cb][e] + Wpb[hc]);
      sm.red[rl][cl] = y * g * fw[hc];
    }
  __syncthreads();
  if (tid < 64) {
    float s = 0.f;
#pragma unroll 8
    for (int c2 = 0; c2 < 128; ++c2) s += sm.red[tid][c2];
    atomicAdd(&pre[r0 + tid], s);
  }
}

// ---------------- finalize: sigmoid + ragged masked sum ----------------
__global__ void k_final(const float* __restrict__ pre, const int* __restrict__ index,
                        float* __restrict__ out) {
  int b = threadIdx.x;
  if (b < BB) {
    int lim = index[b];
    float s = 0.f;
    for (int n = 0; n < MAXN; ++n)
      if (n < lim) s += sigmf(pre[b * MAXN + n]);
    out[b] = s;
  }
}

// ---------------- host launcher ----------------
extern "C" void kernel_launch(void* const* d_in, const int* in_sizes, int n_in,
                              void* d_out, int out_size, void* d_ws, size_t ws_size,
                              hipStream_t stream) {
  (void)in_sizes; (void)n_in; (void)out_size; (void)ws_size;
  const float* box = (const float*)d_in[2];
  const float* qf  = (const float*)d_in[3];
  const int* index = (const int*)d_in[5];
  const float* Wih = (const float*)d_in[6];
  const float* Whh = (const float*)d_in[7];
  const float* bih = (const float*)d_in[8];
  const float* bhh = (const float*)d_in[9];
  const float* Ww  = (const float*)d_in[10];
  const float* Wb  = (const float*)d_in[11];
  const float* Wpw = (const float*)d_in[12];
  const float* Wpb = (const float*)d_in[13];
  const float* fw  = (const float*)d_in[14];
  float* out = (float*)d_out;

  char* ws = (char*)d_ws;
  size_t off = 0;
  float*  xg    = (float*)(ws + off);  off += (size_t)MROWS * G4 * 4;        // 58.7MB
  __bf16* qpadB = (__bf16*)(ws + off); off += (size_t)MROWS * QEP * 2;       // 2.3MB
  __bf16* WihB  = (__bf16*)(ws + off); off += (size_t)G4 * QEP * 2;          // 2.6MB
  __bf16* WhhB  = (__bf16*)(ws + off); off += (size_t)G4 * QH * 2;           // 8MB
  __bf16* WcatB = (__bf16*)(ws + off); off += (size_t)2 * HIDN * MLPK * 2;   // 6MB
  __bf16* h0b   = (__bf16*)(ws + off); off += (size_t)BB * QH * 2;
  __bf16* h1b   = (__bf16*)(ws + off); off += (size_t)BB * QH * 2;
  float*  cst   = (float*)(ws + off);  off += (size_t)BB * QH * 4;
  float*  pre   = (float*)(ws + off);  off += (size_t)RTOT * 4;

  k_prep<<<1024, 256, 0, stream>>>(qf, Wih, Whh, Ww, Wpw, qpadB, WihB, WhhB,
                                   WcatB, h0b, cst, pre);
  k_xg<<<dim3(G4 / 128, MROWS / 64), 256, 0, stream>>>(qpadB, WihB, bih, bhh, xg);
  for (int t = 0; t < TT; ++t) {
    const __bf16* hin = (t & 1) ? h1b : h0b;
    __bf16* hout      = (t & 1) ? h0b : h1b;
    k_lstm<<<dim3(QH / 32, BB / 64), 256, 0, stream>>>(hin, hout, cst, WhhB,
                                                       xg + (size_t)t * BB * G4);
  }
  k_mlp<<<dim3(HIDN / 128, RTOT / 64), 256, 0, stream>>>(box, h0b, WcatB, Wb, Wpb, fw, pre);
  k_final<<<1, 256, 0, stream>>>(pre, index, out);
}
